// ViTSelfAttention_50130858279265
// MI455X (gfx1250) — compile-verified, matrix operand order checked by reading, not run on hardware
//
#include <hip/hip_runtime.h>
#include <hip/hip_bf16.h>

#define B_  16
#define S_  577
#define D_  768
#define H_  12
#define DH_ 64
#define M1_ (B_ * S_)          // 9232 rows of X
#define HEADS_ (B_ * H_)       // 192

typedef _Float16 half4  __attribute__((ext_vector_type(4)));
typedef _Float16 half8  __attribute__((ext_vector_type(8)));
typedef _Float16 half16 __attribute__((ext_vector_type(16)));
typedef float    f32x8  __attribute__((ext_vector_type(8)));

// Build a 16-element f16 A/B fragment from an LDS row.
// CDNA5 16-bit A-matrix layout (16x32): lane L holds row M=L%16; its 16 values
// are K = c0..c0+7 and K = c0+16..c0+23 where c0 = (L/16)*8.  Same pattern for
// B (32x16) with N = L%16, data = a row of the (pre-transposed) operand tile.
static __device__ __forceinline__ half16 load_frag(const _Float16* rowp, int c0) {
    half8 lo = *(const half8*)(rowp + c0);
    half8 hi = *(const half8*)(rowp + c0 + 16);
    return __builtin_shufflevector(lo, hi, 0,1,2,3,4,5,6,7,8,9,10,11,12,13,14,15);
}

// gfx1250 async global->LDS copy (ASYNCcnt-tracked, no VGPR round trip).
static __device__ __forceinline__ void async_b128(void* lds, const void* glb) {
    unsigned l = (unsigned)(unsigned long long)lds;   // generic addr low 32 = LDS offset
    asm volatile("global_load_async_to_lds_b128 %0, %1, off"
                 :: "v"(l), "v"(glb) : "memory");
}
static __device__ __forceinline__ void wait_async0() {
    asm volatile("s_wait_asynccnt 0x0" ::: "memory");
}

// ---------------------------------------------------------------------------
// Kernel 1: fused QKV projection.  out = X @ W^T + b, stored f16 head-major
// as [B, H, S, DH].  blockIdx.z selects q/k/v.  Tile: 128(M) x 128(N), K=32.
// 256 threads = 8 waves; each wave owns a 32x64 sub-tile = 2x4 WMMA tiles.
// ---------------------------------------------------------------------------
__global__ __launch_bounds__(256) void qkv_gemm_kernel(
    const float* __restrict__ x,
    const float* __restrict__ wq, const float* __restrict__ bq,
    const float* __restrict__ wk, const float* __restrict__ bk,
    const float* __restrict__ wv, const float* __restrict__ bv,
    _Float16* __restrict__ qws, _Float16* __restrict__ kws,
    _Float16* __restrict__ vws)
{
    const int z = blockIdx.z;
    const float* w    = (z == 0) ? wq : (z == 1) ? wk : wv;
    const float* bias = (z == 0) ? bq : (z == 1) ? bk : bv;
    _Float16*   outw  = (z == 0) ? qws : (z == 1) ? kws : vws;

    __shared__ __align__(16) _Float16 lx[128 * 40];   // X tile, pitch 40 halfs
    __shared__ __align__(16) _Float16 lw[128 * 40];   // W tile

    const int tid  = threadIdx.x;
    const int lane = tid & 31;
    const int wid  = tid >> 5;          // wave 0..7
    const int wm   = wid & 3;           // 4 waves along M
    const int wn   = wid >> 2;          // 2 waves along N
    const int m0   = blockIdx.x * 128;
    const int n0   = blockIdx.y * 128;

    // Per-thread staging coordinates (same every K-step).
    const int srow = tid >> 3;                // 0..31 base row (x4 strided)
    const int sc4  = (tid & 7) << 2;          // 0..28 col (float4)

    f32x8 acc[2][4];
    #pragma unroll
    for (int i = 0; i < 2; ++i)
        #pragma unroll
        for (int j = 0; j < 4; ++j) acc[i][j] = f32x8{};

    for (int kt = 0; kt < 24; ++kt) {
        const int k0 = kt * 32;

        // Issue ALL global loads first (clamped rows: no exec branches,
        // garbage rows are masked in the epilogue), then convert+store.
        float4 vx[4], vw4[4];
        #pragma unroll
        for (int t = 0; t < 4; ++t) {
            int row = srow + t * 32;
            int mg  = m0 + row; if (mg >= M1_) mg = M1_ - 1;
            vx[t]  = *(const float4*)(x + (size_t)mg * D_ + k0 + sc4);
            vw4[t] = *(const float4*)(w + (size_t)(n0 + row) * D_ + k0 + sc4);
        }
        // Prefetch next K-slab of W (emits global_prefetch_b8).
        __builtin_prefetch(w + (size_t)(n0 + (tid >> 1)) * D_ + (((kt + 1) % 24) * 32), 0, 3);
        #pragma unroll
        for (int t = 0; t < 4; ++t) {
            int row = srow + t * 32;
            half4 hx = { (_Float16)vx[t].x, (_Float16)vx[t].y,
                         (_Float16)vx[t].z, (_Float16)vx[t].w };
            half4 hw = { (_Float16)vw4[t].x, (_Float16)vw4[t].y,
                         (_Float16)vw4[t].z, (_Float16)vw4[t].w };
            *(half4*)(&lx[row * 40 + sc4]) = hx;
            *(half4*)(&lw[row * 40 + sc4]) = hw;
        }
        __syncthreads();

        const int c0 = (lane >> 4) << 3;
        half16 afr[2], bfr[4];
        #pragma unroll
        for (int mi = 0; mi < 2; ++mi)
            afr[mi] = load_frag(&lx[(wm * 32 + mi * 16 + (lane & 15)) * 40], c0);
        #pragma unroll
        for (int ni = 0; ni < 4; ++ni)
            bfr[ni] = load_frag(&lw[(wn * 64 + ni * 16 + (lane & 15)) * 40], c0);
        #pragma unroll
        for (int mi = 0; mi < 2; ++mi)
            #pragma unroll
            for (int ni = 0; ni < 4; ++ni)
                acc[mi][ni] = __builtin_amdgcn_wmma_f32_16x16x32_f16(
                    false, afr[mi], false, bfr[ni], (short)0, acc[mi][ni],
                    false, false);
        __syncthreads();
    }

    // Epilogue: +bias, f32 -> f16, scatter head-major [B,H,S,DH].
    #pragma unroll
    for (int mi = 0; mi < 2; ++mi) {
        #pragma unroll
        for (int ni = 0; ni < 4; ++ni) {
            #pragma unroll
            for (int r = 0; r < 8; ++r) {
                int row_l = wm * 32 + mi * 16 + r + ((lane >> 4) << 3);
                int col_l = wn * 64 + ni * 16 + (lane & 15);
                int mg = m0 + row_l;
                int ng = n0 + col_l;
                if (mg < M1_) {
                    int b = mg / S_;
                    int s = mg - b * S_;
                    int h = ng >> 6;
                    int d = ng & 63;
                    float v = acc[mi][ni][r] + bias[ng];
                    outw[(((size_t)(b * H_ + h) * S_ + s) << 6) + d] = (_Float16)v;
                }
            }
        }
    }
}

// ---------------------------------------------------------------------------
// Kernel 2: attention for one (b,h) and a 16-query tile.
// 64 threads = 2 waves.  scores -> softmax (probs fp32 out) -> ctx = P@V.
// Q/K tiles staged with global_load_async_to_lds_b128 (ASYNCcnt path).
// Static LDS: 2304 + 5120 + 37376 + 19456 = 64256 B (< 64 KB).
// ---------------------------------------------------------------------------
__global__ __launch_bounds__(64) void attention_kernel(
    const _Float16* __restrict__ qws, const _Float16* __restrict__ kws,
    const _Float16* __restrict__ vws,
    float* __restrict__ probs, float* __restrict__ ctx)
{
    __shared__ __align__(16) _Float16 lq[16 * 72];     // Q tile [16 q][64 dh]
    __shared__ __align__(16) _Float16 lkv[2560];       // K tile [32][72] / V^T [64][40]
    __shared__ __align__(16) float    lsc[16 * 584];   // score strip (padded)
    __shared__ __align__(16) _Float16 lpr[16 * 608];   // probs f16 (19 chunks of 32)

    const int tid  = threadIdx.x;
    const int lane = tid & 31;
    const int wid  = tid >> 5;              // 0..1
    const int bh   = blockIdx.y;            // 0..191
    const int q0   = blockIdx.x * 16;
    const size_t base = (size_t)bh * S_ * DH_;
    const float scale = 0.125f;             // 1/sqrt(64)

    // Async-copy Q tile (clamped rows; invalid q rows masked at stores).
    #pragma unroll
    for (int c8 = tid; c8 < 128; c8 += 64) {
        int row = c8 >> 3, coff = (c8 & 7) << 3;
        int q = q0 + row; if (q >= S_) q = S_ - 1;
        async_b128(&lq[row * 72 + coff], qws + base + (size_t)q * DH_ + coff);
    }
    for (int i = tid; i < 16 * 584; i += 64) lsc[i] = -1e30f;
    wait_async0();
    __syncthreads();

    // ---- Phase 1: scores = scale * Q K^T over 19 key-chunks of 32 ----
    for (int ckt = 0; ckt < 19; ++ckt) {
        const int kb = ckt * 32;
        #pragma unroll
        for (int c8 = tid; c8 < 256; c8 += 64) {       // 32*64/8 chunks
            int row = c8 >> 3, coff = (c8 & 7) << 3;
            int key = kb + row; if (key >= S_) key = S_ - 1;  // masked at lsc store
            async_b128(&lkv[row * 72 + coff], kws + base + (size_t)key * DH_ + coff);
        }
        wait_async0();
        __syncthreads();

        f32x8 sacc = f32x8{};
        const int c0 = (lane >> 4) << 3;
        #pragma unroll
        for (int ks = 0; ks < 2; ++ks) {               // DH=64 -> two K=32 steps
            half16 a = load_frag(&lq[(lane & 15) * 72], ks * 32 + c0);
            half16 b = load_frag(&lkv[(wid * 16 + (lane & 15)) * 72], ks * 32 + c0);
            sacc = __builtin_amdgcn_wmma_f32_16x16x32_f16(
                false, a, false, b, (short)0, sacc, false, false);
        }
        #pragma unroll
        for (int r = 0; r < 8; ++r) {
            int row = r + ((lane >> 4) << 3);
            int col = kb + wid * 16 + (lane & 15);
            if (col < S_) lsc[row * 584 + col] = sacc[r] * scale;
        }
        __syncthreads();
    }

    // ---- Phase 2: row softmax (4 threads per row), write probs fp32 ----
    {
        const int r = tid >> 2, g = tid & 3;
        float m = -1e30f;
        for (int j = 0; j < 146; ++j)
            m = fmaxf(m, lsc[r * 584 + g + j * 4]);
        m = fmaxf(m, __shfl_xor(m, 1, 4));
        m = fmaxf(m, __shfl_xor(m, 2, 4));
        float sum = 0.f;
        for (int j = 0; j < 146; ++j) {
            int c = g + j * 4;
            float e = __expf(lsc[r * 584 + c] - m);
            lsc[r * 584 + c] = e;
            sum += e;
        }
        sum += __shfl_xor(sum, 1, 4);
        sum += __shfl_xor(sum, 2, 4);
        const float inv = 1.0f / sum;
        const int q = q0 + r;
        for (int j = 0; j < 152; ++j) {
            int c = g + j * 4;
            float p = (c < 584) ? lsc[r * 584 + c] * inv : 0.f;
            if (c >= S_) p = 0.f;                      // exact zeros beyond S
            lpr[r * 608 + c] = (_Float16)p;
            if (q < S_ && c < S_)
                probs[((size_t)bh * S_ + q) * S_ + c] = p;
        }
    }
    __syncthreads();

    // ---- Phase 3: ctx = P @ V over 19 key-chunks (V staged transposed) ----
    f32x8 cacc[2] = { f32x8{}, f32x8{} };              // di = wid, wid+2
    for (int ckt = 0; ckt < 19; ++ckt) {
        const int kb = ckt * 32;
        for (int e = tid; e < 2048; e += 64) {         // V^T [64 d][32 kk], pitch 40
            int dd = e >> 5, kk = e & 31;
            int key = kb + kk; if (key >= S_) key = S_ - 1;  // p==0 there
            lkv[dd * 40 + kk] = vws[base + (size_t)key * DH_ + dd];
        }
        __syncthreads();
        const int c0 = (lane >> 4) << 3;
        half16 a = load_frag(&lpr[(lane & 15) * 608], kb + c0);
        #pragma unroll
        for (int t = 0; t < 2; ++t) {
            int di = wid + t * 2;
            half16 b = load_frag(&lkv[(di * 16 + (lane & 15)) * 40], c0);
            cacc[t] = __builtin_amdgcn_wmma_f32_16x16x32_f16(
                false, a, false, b, (short)0, cacc[t], false, false);
        }
        __syncthreads();
    }

    const int b_ = bh / H_, h_ = bh - b_ * H_;
    #pragma unroll
    for (int t = 0; t < 2; ++t) {
        int di = wid + t * 2;
        #pragma unroll
        for (int r = 0; r < 8; ++r) {
            int row = r + ((lane >> 4) << 3);
            int q = q0 + row;
            if (q < S_) {
                int d = di * 16 + (lane & 15);
                ctx[((size_t)b_ * S_ + q) * D_ + h_ * DH_ + d] = cacc[t][r];
            }
        }
    }
}

// ---------------------------------------------------------------------------
extern "C" void kernel_launch(void* const* d_in, const int* in_sizes, int n_in,
                              void* d_out, int out_size, void* d_ws, size_t ws_size,
                              hipStream_t stream) {
    (void)in_sizes; (void)n_in; (void)out_size; (void)ws_size;
    const float* x  = (const float*)d_in[0];
    const float* wq = (const float*)d_in[1];
    const float* bq = (const float*)d_in[2];
    const float* wk = (const float*)d_in[3];
    const float* bk = (const float*)d_in[4];
    const float* wv = (const float*)d_in[5];
    const float* bv = (const float*)d_in[6];

    float* ctx   = (float*)d_out;                          // [B,S,D]
    float* probs = ctx + (size_t)B_ * S_ * D_;             // [B,H,S,S]

    const size_t per = (size_t)HEADS_ * S_ * DH_;          // 7,090,176 f16 each
    _Float16* qws = (_Float16*)d_ws;
    _Float16* kws = qws + per;
    _Float16* vws = kws + per;

    // QKV projections: M tiles = ceil(9232/128)=73, N tiles = 768/128 = 6, z = q/k/v
    qkv_gemm_kernel<<<dim3(73, 6, 3), 256, 0, stream>>>(
        x, wq, bq, wk, bk, wv, bv, qws, kws, vws);

    // Attention: 37 query tiles of 16 x 192 (b*h) heads
    attention_kernel<<<dim3(37, HEADS_), 64, 0, stream>>>(
        qws, kws, vws, probs, ctx);
}
